// MetricConv_953482740316
// MI455X (gfx1250) — matrix operand order repaired and checked by reading, not compiled, not run on hardware
//
#include <hip/hip_runtime.h>
#include <hip/hip_bf16.h>
#include <stdint.h>

// ---------------------------------------------------------------------------
// MetricConv for MI455X (gfx1250, wave32, WMMA).
// Pipeline:
//   k_init        : zero out[], s[], init segment-max keys
//   k_cvtT x4     : f32 weights -> transposed f16 (WMMA B-friendly layout)
//   k_node_gemm   : cl = nc@Wl+bl, cr = nc@Wr+br   (v_wmma_f32_16x16x32_f16)
//   k_alpha       : per-edge attention logit (wave-per-edge dot + selu + mask)
//   k_segmax      : segment max over dst via ordered-uint atomicMax
//   k_expsum      : ex = exp(a-m), segment sum via global_atomic_add_f32
//   k_mlp_scatter : fused per-edge 2-layer SELU MLP in WMMA; W1t/W2t filled
//                   into 320KB-class LDS via GLOBAL_LOAD_ASYNC_TO_LDS_B128
//                   (ASYNCcnt) when available; global_prefetch_b8 for next
//                   tile metadata; * softmax weight; atomic scatter-add
//   k_final       : sigmoid(out+bias) / stage_metrics overwrite
// ---------------------------------------------------------------------------

typedef __attribute__((ext_vector_type(16))) _Float16 v16h;
typedef __attribute__((ext_vector_type(8)))  _Float16 v8h;
typedef __attribute__((ext_vector_type(8)))  float    v8f;
typedef __attribute__((vector_size(16)))     int      v4i_g;  // matches builtin

union F16Frag { v16h v; v8h h2[2]; _Float16 h[16]; };

#define N_NODES 100000
#define N_EDGES 1000000
#define CCH     256     // context channels (K of node GEMM, first 256 of MLP K)
#define DMCH    128     // stage_metrics channels
#define HCH     192     // hidden
#define OCH     128     // out channels
#define W1K     384     // CCH + DMCH
#define W1S     392     // padded row stride (halves) of W1t
#define W2S     200     // padded row stride (halves) of W2t
#define MLP_WAVES 8

#if defined(__has_builtin)
#  if __has_builtin(__builtin_amdgcn_global_load_async_to_lds_b128)
#    define USE_ASYNC_LDS 1
#  endif
#endif
#ifndef USE_ASYNC_LDS
#  define USE_ASYNC_LDS 0
#endif

#define AS1 __attribute__((address_space(1)))
#define AS3 __attribute__((address_space(3)))

__device__ __forceinline__ float selu_f(float x) {
  const float sc = 1.0507009873554804934f;
  const float al = 1.6732632423543772848f;
  return sc * (x > 0.f ? x : al * (__expf(x) - 1.f));
}
__device__ __forceinline__ unsigned fkey(float f) {
  unsigned u = __float_as_uint(f);
  return (u >> 31) ? ~u : (u | 0x80000000u);
}
__device__ __forceinline__ float funkey(unsigned k) {
  unsigned u = (k >> 31) ? (k ^ 0x80000000u) : ~k;
  return __uint_as_float(u);
}
__device__ __forceinline__ v8f wmma16(v16h a, v16h b, v8f c) {
  // D = A(16x32 f16) * B(32x16 f16) + C(16x16 f32)
  return __builtin_amdgcn_wmma_f32_16x16x32_f16(false, a, false, b, (short)0, c,
                                                false, false);
}
__device__ __forceinline__ void atomic_add_f32(float* p, float v) {
  __hip_atomic_fetch_add(p, v, __ATOMIC_RELAXED, __HIP_MEMORY_SCOPE_AGENT);
}
__device__ __forceinline__ void wait_asynccnt0() {
#if defined(__has_builtin)
#  if __has_builtin(__builtin_amdgcn_s_wait_asynccnt)
  __builtin_amdgcn_s_wait_asynccnt(0);
  return;
#  endif
#endif
  asm volatile("s_wait_asynccnt 0x0" ::: "memory");
}

// ---------------------------------------------------------------- init ------
__global__ void k_init(float* out, float* sbuf, unsigned* mkey) {
  long stride = (long)gridDim.x * blockDim.x;
  long tot = (long)N_NODES * OCH;
  for (long i = blockIdx.x * (long)blockDim.x + threadIdx.x; i < tot; i += stride) {
    out[i] = 0.f;
    if (i < N_NODES) { sbuf[i] = 0.f; mkey[i] = fkey(-1e30f); }
  }
}

// --------------------------------------------- transpose+convert weights ---
// src: f32 [K][Ncols] row-major  ->  dst: f16 [Ncols][dstride]
__global__ void k_cvtT(const float* __restrict__ src, _Float16* __restrict__ dst,
                       int K, int Ncols, int dstride) {
  int i = blockIdx.x * blockDim.x + threadIdx.x;
  if (i >= K * Ncols) return;
  int k = i / Ncols, n = i % Ncols;
  dst[(size_t)n * dstride + k] = (_Float16)src[i];
}

// ------------------------------------------------------- node transform ----
// One wave per 16-node tile. A-tile staged in per-wave LDS (f16), B fragments
// streamed from transposed f16 weights (L2-resident).
__global__ void __launch_bounds__(256) k_node_gemm(
    const float* __restrict__ ss, const float* __restrict__ se,
    const float* __restrict__ ctx,
    const _Float16* __restrict__ wlt, const _Float16* __restrict__ wrt,
    const float* __restrict__ bl, const float* __restrict__ br,
    _Float16* __restrict__ cl, _Float16* __restrict__ cr) {
  extern __shared__ char smemraw[];
  const int ASTR = 264;  // padded row stride (halves)
  int wave = threadIdx.x >> 5, lane = threadIdx.x & 31;
  _Float16* A = (_Float16*)smemraw + wave * 16 * ASTR;
  int tile = blockIdx.x * (blockDim.x >> 5) + wave;
  if (tile >= N_NODES / 16) return;
  int node0 = tile * 16;

  // Build nc = [ss | ctx | se] tile as f16 in LDS (lane owns 8 channels).
  int c0 = lane * 8;
  for (int m = 0; m < 16; ++m) {
    int node = node0 + m;
#pragma unroll
    for (int j = 0; j < 8; ++j) {
      int c = c0 + j;
      float x = (c < 16)  ? ss[(size_t)node * 16 + c]
              : (c < 240) ? ctx[(size_t)node * 224 + (c - 16)]
                          : se[(size_t)node * 16 + (c - 240)];
      A[m * ASTR + c] = (_Float16)x;
    }
  }
  asm volatile("s_wait_dscnt 0" ::: "memory");  // per-wave LDS store->load

  int nsel = lane & 15;
  int hi8  = (lane >= 16) ? 8 : 0;   // A-layout K offset
  int kbo  = (lane >= 16) ? 16 : 0;  // unused (B loads use hi8)
#pragma unroll
  for (int mat = 0; mat < 2; ++mat) {
    const _Float16* wt = mat ? wrt : wlt;
    const float* bias = mat ? br : bl;
    _Float16* dstm = mat ? cr : cl;
    for (int nt = 0; nt < 16; ++nt) {
      int n = nt * 16 + nsel;
      float bb = bias[n];
      v8f acc;
#pragma unroll
      for (int r = 0; r < 8; ++r) acc[r] = bb;
#pragma unroll
      for (int kt = 0; kt < 8; ++kt) {
        int k0 = kt * 32 + hi8;
        F16Frag a, b;
        a.h2[0] = *(const v8h*)(A + nsel * ASTR + k0);
        a.h2[1] = *(const v8h*)(A + nsel * ASTR + k0 + 16);
        const v8h* bp = (const v8h*)(wt + (size_t)n * CCH + kt * 32 + hi8);
        b.h2[0] = bp[0]; b.h2[1] = bp[2];
        acc = wmma16(a.v, b.v, acc);
      }
#pragma unroll
      for (int r = 0; r < 8; ++r) {
        int m = r + hi8;
        dstm[(size_t)(node0 + m) * CCH + n] = (_Float16)acc[r];
      }
    }
  }
}

// ---------------------------------------------------- attention logits -----
// One wave per edge: alpha = selu(cl[src]+cr[dst]) . att, masked by mj==0.
__global__ void k_alpha(const long long* __restrict__ ei,
                        const _Float16* __restrict__ cl,
                        const _Float16* __restrict__ cr,
                        const float* __restrict__ mj,
                        const float* __restrict__ att,
                        float* __restrict__ alpha) {
  int w = (blockIdx.x * blockDim.x + threadIdx.x) >> 5;
  int lane = threadIdx.x & 31;
  if (w >= N_EDGES) return;
  int s_ = (int)ei[w];
  int d_ = (int)ei[(size_t)N_EDGES + w];
  int c0 = lane * 8;
  v8h a0 = *(const v8h*)(cl + (size_t)s_ * CCH + c0);
  v8h a1 = *(const v8h*)(cr + (size_t)d_ * CCH + c0);
  float dot = 0.f;
#pragma unroll
  for (int j = 0; j < 8; ++j)
    dot += selu_f((float)a0[j] + (float)a1[j]) * att[c0 + j];
  const float4 mv = *((const float4*)(mj + (size_t)s_ * DMCH) + lane);
  int nz = (mv.x != 0.f) | (mv.y != 0.f) | (mv.z != 0.f) | (mv.w != 0.f);
#pragma unroll
  for (int off = 16; off; off >>= 1) {
    dot += __shfl_xor(dot, off, 32);
    nz  |= __shfl_xor(nz, off, 32);
  }
  if (lane == 0) alpha[w] = nz ? dot : 0.f;
}

// -------------------------------------------------------- segment max ------
__global__ void k_segmax(const long long* __restrict__ ei,
                         const float* __restrict__ alpha,
                         unsigned* __restrict__ mkey) {
  int e = blockIdx.x * blockDim.x + threadIdx.x;
  if (e >= N_EDGES) return;
  float a = alpha[e];
  if (a != 0.f) {
    int d = (int)ei[(size_t)N_EDGES + e];
    atomicMax(&mkey[d], fkey(a));
  }
}

// ---------------------------------------------------- exp + segment sum ----
__global__ void k_expsum(const long long* __restrict__ ei,
                         float* __restrict__ exbuf,  // in: alpha, out: ex
                         const unsigned* __restrict__ mkey,
                         float* __restrict__ sbuf) {
  int e = blockIdx.x * blockDim.x + threadIdx.x;
  if (e >= N_EDGES) return;
  float a = exbuf[e];
  float ex = 0.f;
  if (a != 0.f) {
    int d = (int)ei[(size_t)N_EDGES + e];
    ex = __expf(a - funkey(mkey[d]));
    atomic_add_f32(&sbuf[d], ex);
  }
  exbuf[e] = ex;
}

// ------------------------------------- fused edge MLP + softmax + scatter ---
// Persistent blocks; W1t+W2t (f16, ~197KB) in LDS + per-wave 16x192 h tile.
// One wave per 16-edge tile: 144 WMMAs layer1, 48 WMMAs layer2.
__global__ void __launch_bounds__(256) k_mlp_scatter(
    const long long* __restrict__ ei,
    const _Float16* __restrict__ cl, const _Float16* __restrict__ cr,
    const float* __restrict__ mj,
    _Float16* __restrict__ w12t,  // W1t [192][392] then W2t [128][200]
    const float* __restrict__ b1, const float* __restrict__ b2,
    const float* __restrict__ exbuf, const float* __restrict__ sbuf,
    float* __restrict__ out) {
  extern __shared__ char smemraw[];
  _Float16* w1s = (_Float16*)smemraw;        // 192*392 halves
  _Float16* w2s = w1s + 192 * W1S;           // 128*200 halves
  _Float16* hs  = w2s + 128 * W2S;           // MLP_WAVES * 16*200 halves

  // ---- weight fill: async DMA straight into LDS when available ----
  {
    const int nvec = (192 * W1S + 128 * W2S) * 2 / 16;  // 16B packets
#if USE_ASYNC_LDS
    AS1 v4i_g* gsrc = (AS1 v4i_g*)w12t;
    AS3 v4i_g* ldst = (AS3 v4i_g*)smemraw;
    for (int i = threadIdx.x; i < nvec; i += blockDim.x)
      __builtin_amdgcn_global_load_async_to_lds_b128(gsrc + i, ldst + i, 0, 0);
    wait_asynccnt0();
#else
    const uint4* src = (const uint4*)w12t;
    uint4* dst = (uint4*)smemraw;
    for (int i = threadIdx.x; i < nvec; i += blockDim.x) dst[i] = src[i];
#endif
  }
  __syncthreads();

  int wave = threadIdx.x >> 5, lane = threadIdx.x & 31;
  _Float16* hw = hs + wave * 16 * W2S;
  int nsel = lane & 15;
  int hi8  = (lane >= 16) ? 8 : 0;
  int kbo  = (lane >= 16) ? 16 : 0;
  const int ntiles = N_EDGES / 16;

  for (int tile = blockIdx.x * MLP_WAVES + wave; tile < ntiles;
       tile += gridDim.x * MLP_WAVES) {
    int e0 = tile * 16;

    // prefetch next tile's gather metadata (global_prefetch_b8)
    {
      int tnext = tile + gridDim.x * MLP_WAVES;
      if (tnext < ntiles) {
        int en = tnext * 16;
        __builtin_prefetch(&ei[en + nsel], 0, 3);
        __builtin_prefetch(&ei[(size_t)N_EDGES + en + nsel], 0, 3);
        __builtin_prefetch(&exbuf[en + nsel], 0, 3);
      }
    }

    int srcm = (int)ei[e0 + nsel];
    int dstm = (int)ei[(size_t)N_EDGES + e0 + nsel];
    int dste[8]; float avals[8];
#pragma unroll
    for (int r = 0; r < 8; ++r) {
      int e = e0 + hi8 + r;
      int d = (int)ei[(size_t)N_EDGES + e];
      dste[r] = d;
      float ex = exbuf[e];
      avals[r] = (ex != 0.f) ? ex / (sbuf[d] + 1e-16f) : 0.f;
    }

    // ---------------- layer 1: [16x384] x [384x192] ----------------
    v8f acc[12];
#pragma unroll
    for (int nt = 0; nt < 12; ++nt) {
      float bb = b1[nt * 16 + nsel];
#pragma unroll
      for (int r = 0; r < 8; ++r) acc[nt][r] = bb;
    }
    for (int kt = 0; kt < 12; ++kt) {
      F16Frag a;
      int k0 = kt * 32 + hi8;
      if (kt < 8) {  // ctx channels: selu(cl[src]+cr[dst]) built in-register
        v8h x0 = *(const v8h*)(cl + (size_t)srcm * CCH + k0);
        v8h y0 = *(const v8h*)(cr + (size_t)dstm * CCH + k0);
        v8h x1 = *(const v8h*)(cl + (size_t)srcm * CCH + k0 + 16);
        v8h y1 = *(const v8h*)(cr + (size_t)dstm * CCH + k0 + 16);
#pragma unroll
        for (int j = 0; j < 8; ++j) {
          a.h[j]     = (_Float16)selu_f((float)x0[j] + (float)y0[j]);
          a.h[j + 8] = (_Float16)selu_f((float)x1[j] + (float)y1[j]);
        }
      } else {       // stage_metrics channels
        const float* mp = mj + (size_t)srcm * DMCH + (k0 - CCH);
#pragma unroll
        for (int j = 0; j < 8; ++j) {
          a.h[j]     = (_Float16)mp[j];
          a.h[j + 8] = (_Float16)mp[j + 16];
        }
      }
#pragma unroll
      for (int nt = 0; nt < 12; ++nt) {
        const v8h* bp =
            (const v8h*)(w1s + (size_t)(nt * 16 + nsel) * W1S + kt * 32 + hi8);
        F16Frag b; b.h2[0] = bp[0]; b.h2[1] = bp[2];
        acc[nt] = wmma16(a.v, b.v, acc[nt]);
      }
    }
    // h = selu(.) -> per-wave LDS tile (C-layout -> A-layout relayout)
#pragma unroll
    for (int nt = 0; nt < 12; ++nt) {
      int n = nt * 16 + nsel;
#pragma unroll
      for (int r = 0; r < 8; ++r)
        hw[(r + hi8) * W2S + n] = (_Float16)selu_f(acc[nt][r]);
    }
    asm volatile("s_wait_dscnt 0" ::: "memory");

    // ---------------- layer 2: [16x192] x [192x128] ----------------
    v8f acc2[8];
#pragma unroll
    for (int nt = 0; nt < 8; ++nt) {
      float bb = b2[nt * 16 + nsel];
#pragma unroll
      for (int r = 0; r < 8; ++r) acc2[nt][r] = bb;
    }
    for (int kt = 0; kt < 6; ++kt) {
      int k0 = kt * 32 + hi8;
      F16Frag a;
      a.h2[0] = *(const v8h*)(hw + nsel * W2S + k0);
      a.h2[1] = *(const v8h*)(hw + nsel * W2S + k0 + 16);
#pragma unroll
      for (int nt = 0; nt < 8; ++nt) {
        const v8h* bp =
            (const v8h*)(w2s + (size_t)(nt * 16 + nsel) * W2S + kt * 32 + hi8);
        F16Frag b; b.h2[0] = bp[0]; b.h2[1] = bp[2];
        acc2[nt] = wmma16(a.v, b.v, acc2[nt]);
      }
    }
    // msg = selu(f) * a  -> atomic scatter-add into out[dst]
#pragma unroll
    for (int nt = 0; nt < 8; ++nt) {
      int n = nt * 16 + nsel;
#pragma unroll
      for (int r = 0; r < 8; ++r) {
        if (avals[r] != 0.f) {
          float msg = selu_f(acc2[nt][r]) * avals[r];
          atomic_add_f32(&out[(size_t)dste[r] * OCH + n], msg);
        }
      }
    }
  }
}

// ------------------------------------------------------------ finalize -----
__global__ void k_final(float* __restrict__ out, const float* __restrict__ sm,
                        const float* __restrict__ bias) {
  int w = (blockIdx.x * blockDim.x + threadIdx.x) >> 5;
  int lane = threadIdx.x & 31;
  if (w >= N_NODES) return;
  float4* op = (float4*)(out + (size_t)w * OCH) + lane;
  float4 v = *op;
  int nz = (v.x != 0.f) | (v.y != 0.f) | (v.z != 0.f) | (v.w != 0.f);
  nz = __any(nz);
  if (!nz) {
    *op = *((const float4*)(sm + (size_t)w * DMCH) + lane);
  } else {
    float4 b = *((const float4*)bias + lane);
    v.x = 1.f / (1.f + __expf(-(v.x + b.x)));
    v.y = 1.f / (1.f + __expf(-(v.y + b.y)));
    v.z = 1.f / (1.f + __expf(-(v.z + b.z)));
    v.w = 1.f / (1.f + __expf(-(v.w + b.w)));
    *op = v;
  }
}

// ---------------------------------------------------------------------------
extern "C" void kernel_launch(void* const* d_in, const int* in_sizes, int n_in,
                              void* d_out, int out_size, void* d_ws,
                              size_t ws_size, hipStream_t stream) {
  (void)in_sizes; (void)n_in; (void)out_size; (void)ws_size;
  const long long* ei = (const long long*)d_in[0];
  const float* ss   = (const float*)d_in[1];
  const float* se   = (const float*)d_in[2];
  const float* ctx  = (const float*)d_in[3];
  const float* sm   = (const float*)d_in[4];
  const float* W_l  = (const float*)d_in[5];
  const float* b_l  = (const float*)d_in[6];
  const float* W_r  = (const float*)d_in[7];
  const float* b_r  = (const float*)d_in[8];
  const float* att  = (const float*)d_in[9];
  const float* W1   = (const float*)d_in[10];
  const float* b1   = (const float*)d_in[11];
  const float* W2   = (const float*)d_in[12];
  const float* b2   = (const float*)d_in[13];
  const float* bias = (const float*)d_in[14];
  float* out = (float*)d_out;

  char* ws = (char*)d_ws;
  size_t off = 0;
  auto carve = [&](size_t bytes) -> char* {
    char* p = ws + off;
    off = (off + bytes + 255) & ~(size_t)255;
    return p;
  };
  _Float16* cl   = (_Float16*)carve((size_t)N_NODES * CCH * 2);
  _Float16* cr   = (_Float16*)carve((size_t)N_NODES * CCH * 2);
  float*    exb  = (float*)   carve((size_t)N_EDGES * 4);
  float*    sbuf = (float*)   carve((size_t)N_NODES * 4);
  unsigned* mkey = (unsigned*)carve((size_t)N_NODES * 4);
  _Float16* wlt  = (_Float16*)carve((size_t)CCH * CCH * 2);
  _Float16* wrt  = (_Float16*)carve((size_t)CCH * CCH * 2);
  _Float16* w12t = (_Float16*)carve((size_t)(192 * W1S + 128 * W2S) * 2);
  _Float16* w2t  = w12t + 192 * W1S;

  // init out / softmax state
  k_init<<<2048, 256, 0, stream>>>(out, sbuf, mkey);

  // weight transpose+convert to f16
  k_cvtT<<<(CCH * CCH + 255) / 256, 256, 0, stream>>>(W_l, wlt, CCH, CCH, CCH);
  k_cvtT<<<(CCH * CCH + 255) / 256, 256, 0, stream>>>(W_r, wrt, CCH, CCH, CCH);
  k_cvtT<<<(W1K * HCH + 255) / 256, 256, 0, stream>>>(W1, w12t, W1K, HCH, W1S);
  k_cvtT<<<(HCH * OCH + 255) / 256, 256, 0, stream>>>(W2, w2t, HCH, OCH, W2S);

  // node transform (WMMA)
  {
    const int lds = 8 * 16 * 264 * 2;  // 67,584 B
    (void)hipFuncSetAttribute((const void*)k_node_gemm,
                              hipFuncAttributeMaxDynamicSharedMemorySize, lds);
    int blocks = (N_NODES / 16 + 7) / 8;
    k_node_gemm<<<blocks, 256, lds, stream>>>(ss, se, ctx, wlt, wrt, b_l, b_r,
                                              cl, cr);
  }

  // attention logits, segment softmax
  k_alpha<<<(N_EDGES * 32 + 255) / 256, 256, 0, stream>>>(ei, cl, cr, sm, att,
                                                          exb);
  k_segmax<<<(N_EDGES + 255) / 256, 256, 0, stream>>>(ei, exb, mkey);
  k_expsum<<<(N_EDGES + 255) / 256, 256, 0, stream>>>(ei, exb, mkey, sbuf);

  // fused edge MLP + scatter (WMMA, big-LDS weights, persistent blocks)
  {
    const int lds = (192 * W1S + 128 * W2S + MLP_WAVES * 16 * W2S) * 2;  // 252,928 B
    (void)hipFuncSetAttribute((const void*)k_mlp_scatter,
                              hipFuncAttributeMaxDynamicSharedMemorySize, lds);
    k_mlp_scatter<<<1024, 256, lds, stream>>>(ei, cl, cr, sm, w12t, b1, b2, exb,
                                              sbuf, out);
  }

  // finalize
  k_final<<<(N_NODES * 32 + 255) / 256, 256, 0, stream>>>(out, sm, bias);
}
